// GraphCastModel_56676388438029
// MI455X (gfx1250) — matrix operand order repaired and checked by reading, not compiled, not run on hardware
//
#include <hip/hip_runtime.h>
#include <hip/hip_bf16.h>
#include <stdint.h>

typedef __bf16 bf16;
typedef __attribute__((ext_vector_type(16))) __bf16 v16bf;
typedef __attribute__((ext_vector_type(8)))  float  v8f;

#define N_NODES 32768
#define N_EDGES 196608
#define N_LAYERS 8

// ---------------------------------------------------------------------------
// Generic bf16 WMMA GEMM:  C[M][ldc] = act( A[M][lda](bf16) * B[Kp][ldb](bf16) + bias )
// Tile: 128x128 per workgroup (256 threads = 8 waves), K-step 64 (2 wmma K-subs).
// Wave w computes rows [w*16, w*16+16) x all 128 cols (8 wmma tiles, 16 wmma/stage).
// Requirements (guaranteed by caller): M % 128 == 0, Kp % 64 == 0,
// lda/ldb multiples of 16 elements, A/B 16-byte aligned.
// ---------------------------------------------------------------------------
__global__ __launch_bounds__(256)
void wmma_gemm_bf16(const bf16* __restrict__ A, int lda,
                    const bf16* __restrict__ B, int ldb,
                    int Kp, int Np, int Nreal,
                    const float* __restrict__ bias, int act_silu,
                    float* __restrict__ Cf32, bf16* __restrict__ Cbf16, int ldc)
{
    __shared__ __align__(16) bf16 As[128 * 64];   // [row][k]   16 KB
    __shared__ __align__(16) bf16 Bs[128 * 64];   // [n][k]     16 KB (transposed)

    const int tid  = threadIdx.x;
    const int wave = tid >> 5;
    const int lane = tid & 31;
    const int row0 = blockIdx.y * 128;
    const int col0 = blockIdx.x * 128;

    v8f acc[8];
#pragma unroll
    for (int t = 0; t < 8; ++t)
#pragma unroll
        for (int r = 0; r < 8; ++r) acc[t][r] = 0.0f;

    for (int k0 = 0; k0 < Kp; k0 += 64) {
        // ---- stage A tile: 128 rows x 64 k, vectorized b128 copies ----
        {
            int r  = tid >> 1;            // 0..127
            int kh = (tid & 1) * 32;      // 0 or 32
            const uint4* srcp =
                reinterpret_cast<const uint4*>(A + (size_t)(row0 + r) * lda + k0 + kh);
            uint4* dstp = reinterpret_cast<uint4*>(As + r * 64 + kh);
            if (k0 + 64 < Kp) __builtin_prefetch(srcp + 8, 0, 1);
            dstp[0] = srcp[0];
            dstp[1] = srcp[1];
            dstp[2] = srcp[2];
            dstp[3] = srcp[3];
        }
        // ---- stage B tile transposed into Bs[n][k] ----
        // thread t: two consecutive k-rows (kb, kb+1) x 16 contiguous columns.
        {
            int kb = (tid >> 3) * 2;      // 0,2,...,62
            int nc = (tid & 7) * 16;      // 0,16,...,112
            const bf16* s0 = B + (size_t)(k0 + kb) * ldb + col0 + nc;
            const bf16* s1 = s0 + ldb;
            if (col0 + nc + 16 <= Np) {
                const uint4* v0 = reinterpret_cast<const uint4*>(s0);
                const uint4* v1 = reinterpret_cast<const uint4*>(s1);
                uint4 r0a = v0[0], r0b = v0[1];
                uint4 r1a = v1[0], r1b = v1[1];
                const bf16* p0 = reinterpret_cast<const bf16*>(&r0a);
                const bf16* p1 = reinterpret_cast<const bf16*>(&r1a);
#pragma unroll
                for (int i = 0; i < 8; ++i) {
                    bf16* d = Bs + (nc + i) * 64 + kb;
                    d[0] = p0[i]; d[1] = p1[i];
                }
                p0 = reinterpret_cast<const bf16*>(&r0b);
                p1 = reinterpret_cast<const bf16*>(&r1b);
#pragma unroll
                for (int i = 0; i < 8; ++i) {
                    bf16* d = Bs + (nc + 8 + i) * 64 + kb;
                    d[0] = p0[i]; d[1] = p1[i];
                }
            } else {
#pragma unroll
                for (int i = 0; i < 16; ++i) {
                    bool ok = (col0 + nc + i) < Np;
                    bf16* d = Bs + (nc + i) * 64 + kb;
                    d[0] = ok ? s0[i] : (bf16)0.0f;
                    d[1] = ok ? s1[i] : (bf16)0.0f;
                }
            }
        }
        __syncthreads();

#pragma unroll
        for (int ks = 0; ks < 2; ++ks) {
            // ---- A fragment (16x32 bf16, ISA layout) ----
            // lane<16: row=lane,     K = {0..7, 16..23}
            // lane>=16: row=lane-16, K = {8..15, 24..31}
            v16bf afrag;
            {
                int r  = wave * 16 + (lane & 15);
                int kb = (lane >> 4) * 8;
                const bf16* ap = As + r * 64 + ks * 32;
#pragma unroll
                for (int i = 0; i < 8; ++i) afrag[i]     = ap[kb + i];
#pragma unroll
                for (int i = 0; i < 8; ++i) afrag[8 + i] = ap[kb + 16 + i];
            }
            // ---- 8 column tiles ----
#pragma unroll
            for (int t = 0; t < 8; ++t) {
                v16bf bfrag;   // B 32x16: lane half -> K {0..7,16..23} / {8..15,24..31}, N = lane&15
                int n  = t * 16 + (lane & 15);
                int kb = (lane >> 4) * 8;
                const bf16* bp = Bs + n * 64 + ks * 32 + kb;
#pragma unroll
                for (int i = 0; i < 16; ++i) bfrag[i] = bp[(i & 7) + (i >> 3) * 16];

                acc[t] = __builtin_amdgcn_wmma_f32_16x16x32_bf16(
                    false, afrag, false, bfrag, (short)0, acc[t], false, false);
            }
        }
        __syncthreads();
    }

    // ---- epilogue: C layout — lane<16: M=r, N=lane ; lane>=16: M=r+8, N=lane-16
    const int rbase = row0 + wave * 16 + (lane >> 4) * 8;
    const int cl = lane & 15;
#pragma unroll
    for (int t = 0; t < 8; ++t) {
        int col = col0 + t * 16 + cl;
        if (col >= Np) continue;
        float bv = (bias != nullptr && col < Nreal) ? bias[col] : 0.0f;
#pragma unroll
        for (int r = 0; r < 8; ++r) {
            int row = rbase + r;
            float v = acc[t][r] + bv;
            if (act_silu) v = v / (1.0f + __expf(-v));
            size_t off = (size_t)row * ldc + col;
            if (Cf32)  Cf32[off]  = v;
            if (Cbf16) Cbf16[off] = (bf16)v;
        }
    }
}

// ---------------------------------------------------------------------------
// Elementwise / utility kernels
// ---------------------------------------------------------------------------
__global__ void convert_pad_w(const float* __restrict__ src, bf16* __restrict__ dst,
                              int Ks, int Ns, int Kp, int Np, long long total)
{
    long long i = (long long)blockIdx.x * 256 + threadIdx.x;
    if (i >= total) return;
    int k = (int)(i / Np), n = (int)(i % Np);
    dst[i] = (k < Ks && n < Ns) ? (bf16)src[(size_t)k * Ns + n] : (bf16)0.0f;
}

__global__ void build_node_in(const float* __restrict__ x, bf16* __restrict__ out,
                              long long total) // out[N][128], x is (70, N)
{
    long long i = (long long)blockIdx.x * 256 + threadIdx.x;
    if (i >= total) return;
    int n = (int)(i / 128), c = (int)(i % 128);
    out[i] = (c < 70) ? (bf16)x[(size_t)c * N_NODES + n] : (bf16)0.0f;
}

__global__ void build_edge_in(const float* __restrict__ raw, bf16* __restrict__ out,
                              long long total) // out[E][64], raw (E,3)
{
    long long i = (long long)blockIdx.x * 256 + threadIdx.x;
    if (i >= total) return;
    int e = (int)(i / 64), c = (int)(i % 64);
    out[i] = (c < 3) ? (bf16)raw[(size_t)e * 3 + c] : (bf16)0.0f;
}

__global__ void gather_concat_edge(const bf16* __restrict__ node_bf,
                                   const bf16* __restrict__ edge_bf,
                                   const long long* __restrict__ src,
                                   const long long* __restrict__ dst,
                                   bf16* __restrict__ out, long long total) // [E][640]
{
    long long i = (long long)blockIdx.x * 256 + threadIdx.x;
    if (i >= total) return;
    long long e = i / 640;
    int j = (int)(i % 640);
    bf16 v;
    if (j < 256)      v = node_bf[(size_t)src[e] * 256 + j];
    else if (j < 512) v = node_bf[(size_t)dst[e] * 256 + (j - 256)];
    else              v = edge_bf[(size_t)e * 128 + (j - 512)];
    out[i] = v;
}

__global__ void concat_node(const bf16* __restrict__ node_bf,
                            const float* __restrict__ agg,
                            bf16* __restrict__ out, long long total) // [N][384]
{
    long long i = (long long)blockIdx.x * 256 + threadIdx.x;
    if (i >= total) return;
    long long n = i / 384;
    int j = (int)(i % 384);
    out[i] = (j < 256) ? node_bf[(size_t)n * 256 + j]
                       : (bf16)agg[(size_t)n * 128 + (j - 256)];
}

__global__ void zero_f32(float* __restrict__ p, long long total)
{
    long long i = (long long)blockIdx.x * 256 + threadIdx.x;
    if (i < total) p[i] = 0.0f;
}

__global__ void scatter_add_edges(const float* __restrict__ edge,
                                  const long long* __restrict__ dst,
                                  float* __restrict__ agg, long long total) // E*128
{
    long long i = (long long)blockIdx.x * 256 + threadIdx.x;
    if (i >= total) return;
    long long e = i / 128;
    int c = (int)(i % 128);
    atomicAdd(&agg[(size_t)dst[e] * 128 + c], edge[i]);
}

// residual + LayerNorm, one wave per row; updates base in place + bf16 copy
__global__ __launch_bounds__(256)
void ln_residual(float* __restrict__ base, const float* __restrict__ delta,
                 const float* __restrict__ g, const float* __restrict__ bt,
                 int R, int Wd, bf16* __restrict__ out_bf)
{
    int wave = threadIdx.x >> 5;
    int lane = threadIdx.x & 31;
    int row  = blockIdx.x * 8 + wave;
    if (row >= R) return;

    float* bp = base + (size_t)row * Wd;
    const float* dp = delta + (size_t)row * Wd;
    int per = Wd >> 5;  // 4 or 8
    float vals[8];
    float s = 0.0f;
    for (int i = 0; i < per; ++i) {
        int c = lane + i * 32;
        float v = bp[c] + dp[c];
        vals[i] = v; s += v;
    }
    for (int o = 16; o > 0; o >>= 1) s += __shfl_xor(s, o, 32);
    float mean = s / (float)Wd;
    float vs = 0.0f;
    for (int i = 0; i < per; ++i) { float d = vals[i] - mean; vs += d * d; }
    for (int o = 16; o > 0; o >>= 1) vs += __shfl_xor(vs, o, 32);
    float inv = rsqrtf(vs / (float)Wd + 1e-5f);
    for (int i = 0; i < per; ++i) {
        int c = lane + i * 32;
        float o = (vals[i] - mean) * inv * g[c] + bt[c];
        bp[c] = o;
        out_bf[(size_t)row * Wd + c] = (bf16)o;
    }
}

__global__ void write_output(const float* __restrict__ tmp, float* __restrict__ out,
                             long long total) // tmp[N][80] -> out[69][N]
{
    long long i = (long long)blockIdx.x * 256 + threadIdx.x;
    if (i >= total) return;
    long long c = i / N_NODES;
    long long n = i % N_NODES;
    out[i] = tmp[(size_t)n * 80 + c];
}

// ---------------------------------------------------------------------------
// Workspace layout
// ---------------------------------------------------------------------------
constexpr size_t SZ_NODE_F32 = (size_t)N_NODES * 256 * 4;
constexpr size_t SZ_NODE_BF  = (size_t)N_NODES * 256 * 2;
constexpr size_t SZ_EDGE_F32 = (size_t)N_EDGES * 128 * 4;
constexpr size_t SZ_EDGE_BF  = (size_t)N_EDGES * 128 * 2;
constexpr size_t SZ_AGG      = (size_t)N_NODES * 128 * 4;
constexpr size_t SZ_TMP      = (size_t)N_EDGES * 128 * 4;
constexpr size_t SZ_BIGIN    = (size_t)N_EDGES * 640 * 2;
constexpr size_t SZ_HID      = (size_t)N_EDGES * 512 * 2;

constexpr size_t OFF_NODE_F32 = 0;
constexpr size_t OFF_NODE_BF  = OFF_NODE_F32 + SZ_NODE_F32;
constexpr size_t OFF_EDGE_F32 = OFF_NODE_BF  + SZ_NODE_BF;
constexpr size_t OFF_EDGE_BF  = OFF_EDGE_F32 + SZ_EDGE_F32;
constexpr size_t OFF_AGG      = OFF_EDGE_BF  + SZ_EDGE_BF;
constexpr size_t OFF_TMP      = OFF_AGG      + SZ_AGG;
constexpr size_t OFF_BIGIN    = OFF_TMP      + SZ_TMP;
constexpr size_t OFF_HID      = OFF_BIGIN    + SZ_BIGIN;
constexpr size_t OFF_WENC1    = OFF_HID      + SZ_HID;
constexpr size_t OFF_WENC2    = OFF_WENC1  + (size_t)128 * 256 * 2;
constexpr size_t OFF_WEENC1   = OFF_WENC2  + (size_t)256 * 256 * 2;
constexpr size_t OFF_WEENC2   = OFF_WEENC1 + (size_t)64  * 128 * 2;
constexpr size_t OFF_WDEC1    = OFF_WEENC2 + (size_t)128 * 128 * 2;
constexpr size_t OFF_WDEC2    = OFF_WDEC1  + (size_t)256 * 256 * 2;
constexpr size_t OFF_WPE1     = OFF_WDEC2  + (size_t)256 * 80  * 2;
constexpr size_t OFF_WPE2     = OFF_WPE1   + (size_t)N_LAYERS * 640 * 512 * 2;
constexpr size_t OFF_WPN1     = OFF_WPE2   + (size_t)N_LAYERS * 512 * 128 * 2;
constexpr size_t OFF_WPN2     = OFF_WPN1   + (size_t)N_LAYERS * 384 * 512 * 2;

static inline long long cdiv(long long a, long long b) { return (a + b - 1) / b; }

static void launch_gemm(const bf16* A, int lda, const bf16* B, int ldb,
                        int M, int Kp, int Np, int Nreal,
                        const float* bias, int silu,
                        float* Cf, bf16* Cb, int ldc, hipStream_t s)
{
    dim3 grid((unsigned)((Np + 127) / 128), (unsigned)(M / 128));
    wmma_gemm_bf16<<<grid, 256, 0, s>>>(A, lda, B, ldb, Kp, Np, Nreal,
                                        bias, silu, Cf, Cb, ldc);
}

static void launch_conv(const float* src, bf16* dst, int Ks, int Ns, int Kp, int Np,
                        hipStream_t s)
{
    long long tot = (long long)Kp * Np;
    convert_pad_w<<<(unsigned)cdiv(tot, 256), 256, 0, s>>>(src, dst, Ks, Ns, Kp, Np, tot);
}

// ---------------------------------------------------------------------------
extern "C" void kernel_launch(void* const* d_in, const int* in_sizes, int n_in,
                              void* d_out, int out_size, void* d_ws, size_t ws_size,
                              hipStream_t stream)
{
    (void)in_sizes; (void)n_in; (void)out_size; (void)ws_size;
    char* ws = (char*)d_ws;

    float* node_f32 = (float*)(ws + OFF_NODE_F32);
    bf16*  node_bf  = (bf16*) (ws + OFF_NODE_BF);
    float* edge_f32 = (float*)(ws + OFF_EDGE_F32);
    bf16*  edge_bf  = (bf16*) (ws + OFF_EDGE_BF);
    float* agg      = (float*)(ws + OFF_AGG);
    float* tmp      = (float*)(ws + OFF_TMP);
    bf16*  bigin    = (bf16*) (ws + OFF_BIGIN);
    bf16*  hid      = (bf16*) (ws + OFF_HID);

    bf16* w_enc1  = (bf16*)(ws + OFF_WENC1);
    bf16* w_enc2  = (bf16*)(ws + OFF_WENC2);
    bf16* w_eenc1 = (bf16*)(ws + OFF_WEENC1);
    bf16* w_eenc2 = (bf16*)(ws + OFF_WEENC2);
    bf16* w_dec1  = (bf16*)(ws + OFF_WDEC1);
    bf16* w_dec2  = (bf16*)(ws + OFF_WDEC2);
    bf16* w_pe1   = (bf16*)(ws + OFF_WPE1);
    bf16* w_pe2   = (bf16*)(ws + OFF_WPE2);
    bf16* w_pn1   = (bf16*)(ws + OFF_WPN1);
    bf16* w_pn2   = (bf16*)(ws + OFF_WPN2);

    const float*     x    = (const float*)d_in[0];
    const long long* ei   = (const long long*)d_in[1];
    const long long* srcI = ei;
    const long long* dstI = ei + N_EDGES;
    const float* raw_edge = (const float*)d_in[2];
    const float* enc_w1 = (const float*)d_in[3];  const float* enc_b1 = (const float*)d_in[4];
    const float* enc_w2 = (const float*)d_in[5];  const float* enc_b2 = (const float*)d_in[6];
    const float* eenc_w1= (const float*)d_in[7];  const float* eenc_b1= (const float*)d_in[8];
    const float* eenc_w2= (const float*)d_in[9];  const float* eenc_b2= (const float*)d_in[10];
    const float* pe_w1  = (const float*)d_in[11]; const float* pe_b1  = (const float*)d_in[12];
    const float* pe_w2  = (const float*)d_in[13]; const float* pe_b2  = (const float*)d_in[14];
    const float* pe_g   = (const float*)d_in[15]; const float* pe_bt  = (const float*)d_in[16];
    const float* pn_w1  = (const float*)d_in[17]; const float* pn_b1  = (const float*)d_in[18];
    const float* pn_w2  = (const float*)d_in[19]; const float* pn_b2  = (const float*)d_in[20];
    const float* pn_g   = (const float*)d_in[21]; const float* pn_bt  = (const float*)d_in[22];
    const float* dec_w1 = (const float*)d_in[23]; const float* dec_b1 = (const float*)d_in[24];
    const float* dec_w2 = (const float*)d_in[25]; const float* dec_b2 = (const float*)d_in[26];

    // ---- weight conversion (fp32 -> padded bf16, K padded to x64) ----
    launch_conv(enc_w1,  w_enc1,  70, 256, 128,256, stream);
    launch_conv(enc_w2,  w_enc2,  256,256, 256,256, stream);
    launch_conv(eenc_w1, w_eenc1, 3, 128, 64, 128, stream);
    launch_conv(eenc_w2, w_eenc2, 128,128, 128,128, stream);
    launch_conv(dec_w1,  w_dec1,  256,256, 256,256, stream);
    launch_conv(dec_w2,  w_dec2,  256, 69, 256, 80, stream);
    for (int l = 0; l < N_LAYERS; ++l) {
        launch_conv(pe_w1 + (size_t)l*640*512, w_pe1 + (size_t)l*640*512, 640,512,640,512, stream);
        launch_conv(pe_w2 + (size_t)l*512*128, w_pe2 + (size_t)l*512*128, 512,128,512,128, stream);
        launch_conv(pn_w1 + (size_t)l*384*512, w_pn1 + (size_t)l*384*512, 384,512,384,512, stream);
        launch_conv(pn_w2 + (size_t)l*512*256, w_pn2 + (size_t)l*512*256, 512,256,512,256, stream);
    }

    // ---- node encoder ----
    {
        long long tot = (long long)N_NODES * 128;
        build_node_in<<<(unsigned)cdiv(tot,256), 256, 0, stream>>>(x, bigin, tot);
        launch_gemm(bigin, 128, w_enc1, 256, N_NODES, 128, 256, 256,
                    enc_b1, 1, nullptr, hid, 256, stream);
        launch_gemm(hid, 256, w_enc2, 256, N_NODES, 256, 256, 256,
                    enc_b2, 0, node_f32, node_bf, 256, stream);
    }
    // ---- edge encoder ----
    {
        long long tot = (long long)N_EDGES * 64;
        build_edge_in<<<(unsigned)cdiv(tot,256), 256, 0, stream>>>(raw_edge, bigin, tot);
        launch_gemm(bigin, 64, w_eenc1, 128, N_EDGES, 64, 128, 128,
                    eenc_b1, 1, nullptr, hid, 128, stream);
        launch_gemm(hid, 128, w_eenc2, 128, N_EDGES, 128, 128, 128,
                    eenc_b2, 0, edge_f32, edge_bf, 128, stream);
    }

    // ---- processor layers ----
    for (int l = 0; l < N_LAYERS; ++l) {
        // edge update
        long long totE = (long long)N_EDGES * 640;
        gather_concat_edge<<<(unsigned)cdiv(totE,256), 256, 0, stream>>>(
            node_bf, edge_bf, srcI, dstI, bigin, totE);
        launch_gemm(bigin, 640, w_pe1 + (size_t)l*640*512, 512,
                    N_EDGES, 640, 512, 512, pe_b1 + (size_t)l*512, 1,
                    nullptr, hid, 512, stream);
        launch_gemm(hid, 512, w_pe2 + (size_t)l*512*128, 128,
                    N_EDGES, 512, 128, 128, pe_b2 + (size_t)l*128, 0,
                    tmp, nullptr, 128, stream);
        ln_residual<<<(unsigned)cdiv(N_EDGES,8), 256, 0, stream>>>(
            edge_f32, tmp, pe_g + (size_t)l*128, pe_bt + (size_t)l*128,
            N_EDGES, 128, edge_bf);

        // aggregate
        long long totA = (long long)N_NODES * 128;
        zero_f32<<<(unsigned)cdiv(totA,256), 256, 0, stream>>>(agg, totA);
        long long totS = (long long)N_EDGES * 128;
        scatter_add_edges<<<(unsigned)cdiv(totS,256), 256, 0, stream>>>(
            edge_f32, dstI, agg, totS);

        // node update
        long long totN = (long long)N_NODES * 384;
        concat_node<<<(unsigned)cdiv(totN,256), 256, 0, stream>>>(
            node_bf, agg, bigin, totN);
        launch_gemm(bigin, 384, w_pn1 + (size_t)l*384*512, 512,
                    N_NODES, 384, 512, 512, pn_b1 + (size_t)l*512, 1,
                    nullptr, hid, 512, stream);
        launch_gemm(hid, 512, w_pn2 + (size_t)l*512*256, 256,
                    N_NODES, 512, 256, 256, pn_b2 + (size_t)l*256, 0,
                    tmp, nullptr, 256, stream);
        ln_residual<<<(unsigned)cdiv(N_NODES,8), 256, 0, stream>>>(
            node_f32, tmp, pn_g + (size_t)l*256, pn_bt + (size_t)l*256,
            N_NODES, 256, node_bf);
    }

    // ---- decoder ----
    launch_gemm(node_bf, 256, w_dec1, 256, N_NODES, 256, 256, 256,
                dec_b1, 1, nullptr, hid, 256, stream);
    launch_gemm(hid, 256, w_dec2, 80, N_NODES, 256, 80, 69,
                dec_b2, 0, tmp, nullptr, 80, stream);
    long long totO = (long long)69 * N_NODES;
    write_output<<<(unsigned)cdiv(totO,256), 256, 0, stream>>>(tmp, (float*)d_out, totO);
}
